// VTSNE_18640158064833
// MI455X (gfx1250) — matrix operand order repaired and checked by reading, not compiled, not run on hardware
//
#include <hip/hip_runtime.h>
#include <hip/hip_bf16.h>
#include <math.h>

typedef __attribute__((ext_vector_type(16))) _Float16 v16h;
typedef __attribute__((ext_vector_type(8)))  _Float16 v8h;
typedef __attribute__((ext_vector_type(8)))  float    v8f;
typedef __attribute__((ext_vector_type(4)))  unsigned v4u;
typedef __attribute__((ext_vector_type(8)))  int      v8i;
typedef __attribute__((ext_vector_type(4)))  int      v4i;

#define N_POINTS 50000
#define N_TOPICS 256
#define BATCH    1024
#define EPSC     1e-5f
#define NCHUNK   25
#define CHUNK_N  2000
#define TILES_PER_CHUNK 125   // 2000/16
#define GEMM_WAVES 4
#define MTILES   64           // 1024/16

#define TDM_PATH (__has_builtin(__builtin_amdgcn_tensor_load_to_lds) && \
                  __has_builtin(__builtin_amdgcn_s_wait_tensorcnt))

__device__ __forceinline__ unsigned pcg_hash(unsigned v) {
    unsigned state = v * 747796405u + 2891336453u;
    unsigned word = ((state >> ((state >> 28u) + 4u)) ^ state) * 277803737u;
    return (word >> 22u) ^ word;
}

__device__ __forceinline__ float fast_rcp(float x)  { return __builtin_amdgcn_rcpf(x); }
__device__ __forceinline__ float fast_sqrt(float x) { return __builtin_amdgcn_sqrtf(x); }

// ---------------------------------------------------------------------------
// Kernel 1: sample x = mu + eps*exp(lv/2), p = transform(transform(x)) -> f16
// ---------------------------------------------------------------------------
__global__ __launch_bounds__(256) void gen_embed_kernel(
    const float* __restrict__ mu, const float* __restrict__ lv,
    const int* __restrict__ idx, int nrows, unsigned seed,
    _Float16* __restrict__ outP, float* __restrict__ outS,
    float* __restrict__ kldrow)
{
    const int wave = (int)((blockIdx.x * blockDim.x + threadIdx.x) >> 5);
    const int lane = (int)(threadIdx.x & 31);
    if (wave >= nrows) return;
    const int src = idx ? idx[wave] : wave;
    const int t0  = lane * 8;
    const float* murow = mu + (size_t)src * N_TOPICS + t0;
    const float* lvrow = lv + (size_t)src * N_TOPICS + t0;

    float m[8], l[8], x[8];
#pragma unroll
    for (int q = 0; q < 2; ++q) {
        float4 am = ((const float4*)murow)[q];
        float4 al = ((const float4*)lvrow)[q];
        m[4*q+0]=am.x; m[4*q+1]=am.y; m[4*q+2]=am.z; m[4*q+3]=am.w;
        l[4*q+0]=al.x; l[4*q+1]=al.y; l[4*q+2]=al.z; l[4*q+3]=al.w;
    }

    const unsigned base = (unsigned)wave * (unsigned)N_TOPICS + (unsigned)t0;
    float sq = 0.0f, kacc = 0.0f;
#pragma unroll
    for (int k = 0; k < 8; k += 2) {
        unsigned h1 = pcg_hash(seed ^ (base + (unsigned)k));
        unsigned h2 = pcg_hash(h1 ^ 0x9E3779B9u);
        float u1 = ((float)h1 + 1.0f) * 2.3283064365386963e-10f;  // (0,1]
        float u2 = (float)h2 * 2.3283064365386963e-10f;           // [0,1)
        float rr = fast_sqrt(fmaxf(-2.0f * __logf(u1), 0.0f));
        float ang = 6.283185307179586f * u2;
        float z0 = rr * __cosf(ang);
        float z1 = rr * __sinf(ang);
        x[k]   = fmaf(z0, __expf(0.5f * l[k]),   m[k]);
        x[k+1] = fmaf(z1, __expf(0.5f * l[k+1]), m[k+1]);
    }
#pragma unroll
    for (int k = 0; k < 8; ++k) {
        sq   += x[k] * x[k];
        kacc += 1.0f + l[k] - m[k] * m[k] - __expf(l[k]);
    }
#pragma unroll
    for (int msk = 1; msk < 32; msk <<= 1) {
        sq   += __shfl_xor(sq,   msk, 32);
        kacc += __shfl_xor(kacc, msk, 32);
    }
    float d1 = 1.0f + fast_sqrt(1.0f + sq);
    float sq2 = sq * fast_rcp(d1 * d1);
    float d2 = 1.0f + fast_sqrt(1.0f + sq2);
    float scale = fast_rcp(d1 * d2);

    v8h pv;
#pragma unroll
    for (int k = 0; k < 8; ++k) pv[k] = (_Float16)(x[k] * scale);
    *(v8h*)(outP + (size_t)wave * N_TOPICS + t0) = pv;

    if (lane == 0) {
        float sp = sq * scale * scale;
        outS[wave] = fminf(sp, 1.0f - EPSC);
        if (kldrow) kldrow[wave] = kacc;
    }
}

#if TDM_PATH
// Issue one TDM 2-D tile load: 16 rows x 512B (data_size=8B, 64 units/row)
// into LDS at byte offset lds_off. Wave-uniform args -> SGPR descriptor.
__device__ __forceinline__ void tdm_load_tile_b(const _Float16* gsrc, unsigned lds_off)
{
    unsigned long long ga = (unsigned long long)(uintptr_t)gsrc;
    v4u g0;
    g0[0] = 1u;                                        // count=1, user mode
    g0[1] = lds_off;                                   // lds_addr (bytes)
    g0[2] = (unsigned)(ga & 0xFFFFFFFFull);            // global_addr[31:0]
    g0[3] = (unsigned)((ga >> 32) & 0x01FFFFFFull)     // global_addr[56:32]
          | 0x80000000u;                               // type=2 (bits 127:126)
    v8i g1;
    g1[0] = (int)(3u << 16);      // workgroup_mask=0, data_size=3 (8B)
    g1[1] = (int)(64u << 16);     // tensor_dim0 = 64 (bits 79:48, low half)
    g1[2] = (int)(0xC350u << 16); // tensor_dim1 = 50000 (bits 111:80, low half)
    g1[3] = (int)(64u << 16);     // tensor_dim1 hi=0 | tile_dim0 = 64
    g1[4] = 16;                   // tile_dim1 = 16 rows, tile_dim2 = 0
    g1[5] = 64;                   // tensor_dim0_stride = 64 (512B row pitch)
    g1[6] = 0;
    g1[7] = 0;
    v4i z4 = {0, 0, 0, 0};
    v8i z8 = {0, 0, 0, 0, 0, 0, 0, 0};
    // clang-23 / therock-10.0 form: (g0, g1, g2, g3, extra, cpol)
    __builtin_amdgcn_tensor_load_to_lds(g0, g1, z4, z4, z8, 0);
}
#endif

// ---------------------------------------------------------------------------
// Kernel 2: fused WMMA GEMM (dik = pi @ p^T) + Poincare distance + online
// logsumexp over N. Block = 4 waves; TDM double-buffers B tiles in LDS
// (one tensor_load_to_lds in flight while the current tile is consumed).
// ---------------------------------------------------------------------------
__global__ __launch_bounds__(128) void poincare_gemm_lse_kernel(
    const _Float16* __restrict__ P, const _Float16* __restrict__ PI,
    const float* __restrict__ SV, const float* __restrict__ SU,
    float* __restrict__ PART)
{
    const int lane  = (int)(threadIdx.x & 31);
    const int wave  = (int)(threadIdx.x >> 5);
    const int mtile = (int)blockIdx.x;
    const int chunk = (int)blockIdx.y;
    const int l15 = lane & 15;
    const int hi  = lane >> 4;           // 0 or 1
    const int m0  = mtile * 16;

    // A fragments (pi tile), contiguous k order: k = 16*hi + e
    v16h afrag[8];
    const _Float16* Arow = PI + (size_t)(m0 + l15) * N_TOPICS;
#pragma unroll
    for (int kc = 0; kc < 8; ++kc) {
        const _Float16* pa = Arow + kc * 32 + hi * 16;
        v8h lo  = *(const v8h*)(pa);
        v8h hi8 = *(const v8h*)(pa + 8);
        v16h f;
#pragma unroll
        for (int e = 0; e < 8; ++e) { f[e] = lo[e]; f[8 + e] = hi8[e]; }
        afrag[kc] = f;
    }

    float su_r[8], invu[8], run_m[8], run_s[8];
#pragma unroll
    for (int r = 0; r < 8; ++r) {
        float s = SU[m0 + r + 8 * hi];
        su_r[r] = s;
        invu[r] = fast_rcp(1.0f - s);
        run_m[r] = -3.0e38f;
        run_s[r] = 0.0f;
    }

#if TDM_PATH
    // per-wave double-buffered B tile: 4 waves x 2 bufs x 8KB = 64KB LDS
    __shared__ _Float16 smem[GEMM_WAVES][2][16 * N_TOPICS];
    {   // make LDS "written" in the compiler's view (TDM writes are invisible)
        v8h z = {};
        _Float16* flat = &smem[0][0][0];
        for (int k = (int)threadIdx.x * 8; k < GEMM_WAVES * 2 * 16 * N_TOPICS;
             k += (int)blockDim.x * 8)
            *(v8h*)(flat + k) = z;
        __syncthreads();
    }
    const unsigned lds_base = (unsigned)(uintptr_t)(void*)&smem[wave][0][0];
    const unsigned buf_bytes = 16u * N_TOPICS * 2u;

    // contiguous tile range per wave: 31/31/31/32 of 125
    const int t_lo = (TILES_PER_CHUNK * wave) / GEMM_WAVES;
    const int t_hi = (TILES_PER_CHUNK * (wave + 1)) / GEMM_WAVES;

    tdm_load_tile_b(P + (size_t)(chunk * CHUNK_N + t_lo * 16) * N_TOPICS, lds_base);

    for (int t = t_lo; t < t_hi; ++t) {
        const int cur = (t - t_lo) & 1;
        const int n0 = chunk * CHUNK_N + t * 16;
        if (t + 1 < t_hi) {
            tdm_load_tile_b(P + (size_t)(chunk * CHUNK_N + (t + 1) * 16) * N_TOPICS,
                            lds_base + (unsigned)(cur ^ 1) * buf_bytes);
            __builtin_amdgcn_s_wait_tensorcnt(1);  // current tile complete, next in flight
        } else {
            __builtin_amdgcn_s_wait_tensorcnt(0);
        }
        asm volatile("" ::: "memory");

        const _Float16* lrow = &smem[wave][cur][0] + (size_t)l15 * N_TOPICS + hi * 16;
        v16h bfrag[8];
#pragma unroll
        for (int kc = 0; kc < 8; ++kc)
            bfrag[kc] = *(const v16h*)(lrow + kc * 32);

        v8f c = {0.0f,0.0f,0.0f,0.0f,0.0f,0.0f,0.0f,0.0f};
#pragma unroll
        for (int kc = 0; kc < 8; ++kc)
            c = __builtin_amdgcn_wmma_f32_16x16x32_f16(
                    false, afrag[kc], false, bfrag[kc], (short)0, c, false, false);

        const float sv_n = SV[n0 + l15];
        const float invv = fast_rcp(1.0f - sv_n);
#pragma unroll
        for (int r = 0; r < 8; ++r) {
            float sqd = fmaxf(su_r[r] + sv_n - 2.0f * c[r], 0.0f);
            float xx = fmaf(2.0f * sqd, invu[r] * invv, 1.0f);
            xx = fmaxf(xx, 1.0f + 1e-7f);
            float d = __logf(xx + fast_sqrt(xx * xx - 1.0f));
            float nm = fmaxf(run_m[r], d);
            run_s[r] = run_s[r] * __expf(run_m[r] - nm) + __expf(d - nm);
            run_m[r] = nm;
        }
    }
#else
    // fallback: direct global loads of B fragments
    for (int t = wave; t < TILES_PER_CHUNK; t += GEMM_WAVES) {
        const int n0 = chunk * CHUNK_N + t * 16;
        const _Float16* Brow = P + (size_t)(n0 + l15) * N_TOPICS + hi * 16;
        const int tn = t + GEMM_WAVES;
        if (tn < TILES_PER_CHUNK) {
            const _Float16* nb = P + (size_t)(chunk * CHUNK_N + tn * 16 + l15) * N_TOPICS;
            __builtin_prefetch(nb, 0, 1);
            __builtin_prefetch(nb + 128, 0, 1);
        }
        v16h bfrag[8];
#pragma unroll
        for (int kc = 0; kc < 8; ++kc)
            bfrag[kc] = *(const v16h*)(Brow + kc * 32);
        v8f c = {0.0f,0.0f,0.0f,0.0f,0.0f,0.0f,0.0f,0.0f};
#pragma unroll
        for (int kc = 0; kc < 8; ++kc)
            c = __builtin_amdgcn_wmma_f32_16x16x32_f16(
                    false, afrag[kc], false, bfrag[kc], (short)0, c, false, false);
        const float sv_n = SV[n0 + l15];
        const float invv = fast_rcp(1.0f - sv_n);
#pragma unroll
        for (int r = 0; r < 8; ++r) {
            float sqd = fmaxf(su_r[r] + sv_n - 2.0f * c[r], 0.0f);
            float xx = fmaf(2.0f * sqd, invu[r] * invv, 1.0f);
            xx = fmaxf(xx, 1.0f + 1e-7f);
            float d = __logf(xx + fast_sqrt(xx * xx - 1.0f));
            float nm = fmaxf(run_m[r], d);
            run_s[r] = run_s[r] * __expf(run_m[r] - nm) + __expf(d - nm);
            run_m[r] = nm;
        }
    }
#endif

    // one butterfly merge across the 16 lanes that share each row
#pragma unroll
    for (int r = 0; r < 8; ++r) {
        float M = run_m[r], S = run_s[r];
#pragma unroll
        for (int msk = 1; msk < 16; msk <<= 1) {
            float Mo = __shfl_xor(M, msk, 32);
            float So = __shfl_xor(S, msk, 32);
            float nm = fmaxf(M, Mo);
            S = S * __expf(M - nm) + So * __expf(Mo - nm);
            M = nm;
        }
        run_m[r] = M; run_s[r] = S;
    }

    if (l15 == 0) {
#pragma unroll
        for (int r = 0; r < 8; ++r) {
            const int row = r + 8 * hi;
            const size_t pidx =
                ((((size_t)mtile * NCHUNK + chunk) * GEMM_WAVES + wave) * 16 + row);
            PART[2 * pidx]     = run_m[r];
            PART[2 * pidx + 1] = run_s[r];
        }
    }
}

// ---------------------------------------------------------------------------
// Kernel 3: d_{b, j[b]} — one wave per batch row, f32 dot over f16 vectors
// ---------------------------------------------------------------------------
__global__ __launch_bounds__(128) void dbj_kernel(
    const _Float16* __restrict__ P, const _Float16* __restrict__ PI,
    const float* __restrict__ SV, const float* __restrict__ SU,
    const int* __restrict__ jidx, float* __restrict__ DBJ)
{
    const int b    = (int)((blockIdx.x * blockDim.x + threadIdx.x) >> 5);
    const int lane = (int)(threadIdx.x & 31);
    if (b >= BATCH) return;
    const int jj = jidx[b];
    v8h va = *(const v8h*)(PI + (size_t)b  * N_TOPICS + lane * 8);
    v8h vb = *(const v8h*)(P  + (size_t)jj * N_TOPICS + lane * 8);
    float dot = 0.0f;
#pragma unroll
    for (int e = 0; e < 8; ++e) dot += (float)va[e] * (float)vb[e];
#pragma unroll
    for (int msk = 1; msk < 32; msk <<= 1) dot += __shfl_xor(dot, msk, 32);
    if (lane == 0) {
        float su = SU[b], sv = SV[jj];
        float sqd = fmaxf(su + sv - 2.0f * dot, 0.0f);
        float xx = fmaf(2.0f * sqd, fast_rcp((1.0f - su) * (1.0f - sv)), 1.0f);
        xx = fmaxf(xx, 1.0f + 1e-7f);
        DBJ[b] = __logf(xx + fast_sqrt(xx * xx - 1.0f));
    }
}

// ---------------------------------------------------------------------------
// Kernel 4: finalize — combine LSE partials, compute kld sum, add kld_rp*frac
// ---------------------------------------------------------------------------
__global__ __launch_bounds__(1024) void finalize_kernel(
    const float* __restrict__ KLDROW, const float* __restrict__ PART,
    const float* __restrict__ DBJ, const float* __restrict__ pij,
    float* __restrict__ out)
{
    __shared__ float sdata[1024];
    __shared__ float kr_total;
    const int tid = (int)threadIdx.x;

    float acc = 0.0f;
    for (int idx = tid; idx < N_POINTS; idx += 1024) acc += KLDROW[idx];
    sdata[tid] = acc;
    __syncthreads();
    for (int s = 512; s > 0; s >>= 1) {
        if (tid < s) sdata[tid] += sdata[tid + s];
        __syncthreads();
    }
    if (tid == 0) kr_total = sdata[0];
    __syncthreads();

    const int b = tid;
    const int mtile = b >> 4;
    const int row   = b & 15;
    float M = -3.0e38f, S = 0.0f;
    for (int chunk = 0; chunk < NCHUNK; ++chunk) {
        for (int w = 0; w < GEMM_WAVES; ++w) {
            const size_t pidx =
                ((((size_t)mtile * NCHUNK + chunk) * GEMM_WAVES + w) * 16 + row);
            float pm = PART[2 * pidx], ps = PART[2 * pidx + 1];
            float nm = fmaxf(M, pm);
            S = S * __expf(M - nm) + ps * __expf(pm - nm);
            M = nm;
        }
    }
    float lse = M + __logf(S);
    float pb = pij[b];
    float kld_b = pb * (__logf(pb) - (DBJ[b] - lse));
    __syncthreads();
    sdata[tid] = kld_b;
    __syncthreads();
    for (int s = 512; s > 0; s >>= 1) {
        if (tid < s) sdata[tid] += sdata[tid + s];
        __syncthreads();
    }
    if (tid == 0) {
        const float frac = (float)BATCH / ((float)N_POINTS * (float)N_POINTS);
        out[0] = sdata[0] + (-0.5f * kr_total) * frac;
    }
}

// ---------------------------------------------------------------------------
extern "C" void kernel_launch(void* const* d_in, const int* in_sizes, int n_in,
                              void* d_out, int out_size, void* d_ws, size_t ws_size,
                              hipStream_t stream)
{
    (void)in_sizes; (void)n_in; (void)out_size; (void)ws_size;
    const float* pij = (const float*)d_in[0];
    const float* mu  = (const float*)d_in[1];
    const float* lv  = (const float*)d_in[2];
    const int* i_idx = (const int*)d_in[3];
    const int* j_idx = (const int*)d_in[4];
    float* out = (float*)d_out;

    char* ws = (char*)d_ws;
    size_t off = 0;
    _Float16* P  = (_Float16*)(ws + off); off += (size_t)N_POINTS * N_TOPICS * 2; // 25.6 MB
    _Float16* PIV= (_Float16*)(ws + off); off += (size_t)BATCH    * N_TOPICS * 2; // 0.5 MB
    float* SV     = (float*)(ws + off);   off += (size_t)N_POINTS * 4;
    float* SU     = (float*)(ws + off);   off += (size_t)BATCH * 4;
    float* KLDROW = (float*)(ws + off);   off += (size_t)N_POINTS * 4;
    float* PART   = (float*)(ws + off);   off += (size_t)MTILES * NCHUNK * GEMM_WAVES * 16 * 2 * 4;
    float* DBJ    = (float*)(ws + off);   off += (size_t)BATCH * 4;

    gen_embed_kernel<<<N_POINTS / 8, 256, 0, stream>>>(
        mu, lv, nullptr, N_POINTS, 0x1234567u, P, SV, KLDROW);
    gen_embed_kernel<<<BATCH / 8, 256, 0, stream>>>(
        mu, lv, i_idx, BATCH, 0x89ABCDEu, PIV, SU, nullptr);

    dim3 grid(MTILES, NCHUNK);
    poincare_gemm_lse_kernel<<<grid, 128, 0, stream>>>(P, PIV, SV, SU, PART);

    dbj_kernel<<<BATCH / 4, 128, 0, stream>>>(P, PIV, SV, SU, j_idx, DBJ);

    finalize_kernel<<<1, 1024, 0, stream>>>(KLDROW, PART, DBJ, pij, out);
}
